// LRP_MSDeformAttn_62457414418499
// MI455X (gfx1250) — matrix operand order repaired and checked by reading, not compiled, not run on hardware
//
#include <hip/hip_runtime.h>
#include <hip/hip_bf16.h>

// MSDeformAttn forward for MI455X (gfx1250, wave32, WMMA).
//
//   0) one-shot f32->bf16 conversion of activations + weights (vectorized)
//   1) value = x @ W_val^T + b_val            (WMMA bf16 GEMM, M=39894,N=256,K=256)
//   2) offr  = query @ W_samp^T + b_samp      (WMMA bf16 GEMM, N=256)
//   3) attr  = query @ W_attn^T + b_attn      (WMMA bf16 GEMM, N=128)
//   4) sampling: softmax(16) + bilinear gather, 1 wave per (n,q,h), lane = dh
//      (DH==32==wave32); accumulator stored bf16 for the output GEMM
//   5) out   = acc @ W_out^T + b_out          (WMMA bf16 GEMM, N=256) -> d_out
//
// Roofline: ~18 GFLOP vs ~300 MB moved -> bandwidth-bound at 23.3 TB/s. GEMM
// waves compute 16x64 strips (A fragment reused across 4 WMMAs); operands are
// pre-converted to bf16 so the hot loop is pure b128/b64 loads + v_wmma, with
// no per-element conversion VALU. W matrices (<=256 KB) stay L2-resident.

#define BATCH 2
#define DMODEL 256
#define NHEADS 8
#define NLEVELS 4
#define NPOINTS 4
#define DHEAD 32
#define NTW 4   // 16x16 n-tiles per wave -> 16x64 strip

typedef __attribute__((ext_vector_type(16))) __bf16 v16bf;
typedef __attribute__((ext_vector_type(8)))  __bf16 v8bf;
typedef __attribute__((ext_vector_type(8)))  float  v8f;

__device__ __forceinline__ __bf16 f2bf(float x) {
  // round-to-nearest-even f32 -> bf16 (bit-exact, no dependence on cvt lowering)
  unsigned u = __builtin_bit_cast(unsigned, x);
  unsigned r = u + 0x7FFFu + ((u >> 16) & 1u);
  unsigned short h = (unsigned short)(r >> 16);
  return __builtin_bit_cast(__bf16, h);
}

// Vectorized one-shot conversion: 8 elements per thread (b128+b128 in, b128 out).
__global__ void cvt_f32_bf16(const float* __restrict__ src,
                             __bf16* __restrict__ dst, int n8) {
  const int i = blockIdx.x * blockDim.x + threadIdx.x;
  if (i >= n8) return;
  const size_t idx = (size_t)i * 8;
  const float4 x0 = *(const float4*)(src + idx);
  const float4 x1 = *(const float4*)(src + idx + 4);
  v8bf o;
  o[0] = f2bf(x0.x); o[1] = f2bf(x0.y); o[2] = f2bf(x0.z); o[3] = f2bf(x0.w);
  o[4] = f2bf(x1.x); o[5] = f2bf(x1.y); o[6] = f2bf(x1.z); o[7] = f2bf(x1.w);
  *(v8bf*)(dst + idx) = o;
}

// Y[M x Nout] = X[M x 256] * W[Nout x 256]^T + bias[Nout], X/W pre-converted bf16.
// One wave computes a 16x64 strip: 32 x v_wmma_f32_16x16x32_bf16, zero cvt VALU.
__global__ void gemm256_bf16_wmma(const __bf16* __restrict__ X,
                                  const __bf16* __restrict__ W,
                                  const float* __restrict__ bias,
                                  float* __restrict__ Y,
                                  int M, int Nout) {
  constexpr int K = 256;
  const int lane = threadIdx.x & 31;
  const int wave = blockIdx.x * (blockDim.x >> 5) + (threadIdx.x >> 5);
  const int Mt = (M + 15) >> 4;
  const int Ng = Nout >> 6;                    // 64-column groups
  if (wave >= Mt * Ng) return;                 // wave-uniform: EXEC stays all-ones
  const int tm = wave / Ng;
  const int tg = wave - tm * Ng;

  // A row for this lane; clamp OOB rows to row 0 (their D rows are never stored)
  const int rA = tm * 16 + (lane & 15);
  const __bf16* pa = X + (size_t)(rA < M ? rA : 0) * K;
  // ISA VGPR layouts (CDNA5 ISA §7.12.2):
  //   A 16x32 bf16: lanes 0-15 K=[0..7 | 16..23], lanes 16-31 K=[8..15 | 24..31]
  //   B 32x16 bf16: lanes 0-15 K=[0..7 | 16..23], lanes 16-31 K=[8..15 | 24..31]
  const int aoff = (lane & 16) ? 8 : 0;
  const int boff = (lane & 16) ? 16 : 0;

  const __bf16* pb[NTW];
#pragma unroll
  for (int t = 0; t < NTW; ++t)
    pb[t] = W + (size_t)(tg * 64 + t * 16 + (lane & 15)) * K;

  __builtin_prefetch(pa, 0, 0);                // global_prefetch_b8 on the A stream

  v8f c[NTW] = {};
#pragma unroll
  for (int kc = 0; kc < K; kc += 32) {
    const v8bf alo = *(const v8bf*)(pa + kc + aoff);
    const v8bf ahi = *(const v8bf*)(pa + kc + 16 + aoff);
    const v16bf a = __builtin_shufflevector(alo, ahi, 0, 1, 2, 3, 4, 5, 6, 7,
                                            8, 9, 10, 11, 12, 13, 14, 15);
#pragma unroll
    for (int t = 0; t < NTW; ++t) {
      const v8bf blo = *(const v8bf*)(pb[t] + kc + aoff);
      const v8bf bhi = *(const v8bf*)(pb[t] + kc + 16 + aoff);
      const v16bf b = __builtin_shufflevector(blo, bhi, 0, 1, 2, 3, 4, 5, 6, 7,
                                              8, 9, 10, 11, 12, 13, 14, 15);
      c[t] = __builtin_amdgcn_wmma_f32_16x16x32_bf16(false, a, false, b,
                                                     (short)0, c[t], false, false);
    }
  }

  // C/D layout: VGPR r -> M = r (lanes 0-15) / r+8 (lanes 16-31); N = lane%16
  const int mb = tm * 16 + ((lane & 16) ? 8 : 0);
#pragma unroll
  for (int t = 0; t < NTW; ++t) {
    const int nc = tg * 64 + t * 16 + (lane & 15);
    const float bn = bias[nc];
#pragma unroll
    for (int r = 0; r < 8; ++r) {
      const int m = mb + r;
      if (m < M) Y[(size_t)m * Nout + nc] = c[t][r] + bn;
    }
  }
}

// One wave per (n, q, h); lane = dh channel (DHEAD == 32 == wave32).
__global__ void msda_sample(const float* __restrict__ refpts,   // [N,Lq,L,2]
                            const int*   __restrict__ shapes,   // [L,2] (H,W)
                            const int*   __restrict__ lsi,      // [L]
                            const float* __restrict__ value,    // [N*Lq, 256] col=(h*32+dh)
                            const float* __restrict__ offr,     // [N*Lq, 256]
                            const float* __restrict__ attr,     // [N*Lq, 128]
                            __bf16*      __restrict__ acc,      // [N*Lq, 256] bf16
                            int Lq) {
  const int lane = threadIdx.x & 31;
  const int wave = blockIdx.x * (blockDim.x >> 5) + (threadIdx.x >> 5);
  const int total = BATCH * Lq * NHEADS;
  if (wave >= total) return;
  const int h  = wave & (NHEADS - 1);
  const int nq = wave >> 3;                 // n*Lq + q
  const int n  = nq / Lq;

  // softmax over the 16 (level,point) logits for this head
  const float* tl = attr + (size_t)nq * (NHEADS * NLEVELS * NPOINTS) + h * 16;
  float mx = tl[0];
#pragma unroll
  for (int j = 1; j < 16; ++j) mx = fmaxf(mx, tl[j]);
  float w[16];
  float se = 0.0f;
#pragma unroll
  for (int j = 0; j < 16; ++j) { w[j] = __expf(tl[j] - mx); se += w[j]; }
  const float inv = 1.0f / se;

  const float* offp = offr + (size_t)nq * DMODEL + h * (NLEVELS * NPOINTS * 2);
  const float* refp = refpts + (size_t)nq * (NLEVELS * 2);

  float a = 0.0f;
#pragma unroll
  for (int l = 0; l < NLEVELS; ++l) {
    const int Hl = shapes[2 * l + 0];
    const int Wl = shapes[2 * l + 1];
    const int st = lsi[l];
    const float rx = refp[2 * l + 0];
    const float ry = refp[2 * l + 1];
    const float* vbase = value + ((size_t)n * Lq + st) * DMODEL + h * DHEAD + lane;
#pragma unroll
    for (int p = 0; p < NPOINTS; ++p) {
      const int oc = (l * NPOINTS + p) * 2;
      // loc = ref + off / [W,H];  x = loc_x*W - 0.5, y = loc_y*H - 0.5
      const float x = (rx + offp[oc + 0] / (float)Wl) * (float)Wl - 0.5f;
      const float y = (ry + offp[oc + 1] / (float)Hl) * (float)Hl - 0.5f;
      const float xf = floorf(x), yf = floorf(y);
      const int x0 = (int)xf, y0 = (int)yf;
      const float lx = x - xf, ly = y - yf;
      float s = 0.0f;
      if (x0 >= 0 && x0 < Wl && y0 >= 0 && y0 < Hl)
        s += (1.0f - lx) * (1.0f - ly) * vbase[(size_t)(y0 * Wl + x0) * DMODEL];
      if (x0 + 1 >= 0 && x0 + 1 < Wl && y0 >= 0 && y0 < Hl)
        s += lx * (1.0f - ly) * vbase[(size_t)(y0 * Wl + x0 + 1) * DMODEL];
      if (x0 >= 0 && x0 < Wl && y0 + 1 >= 0 && y0 + 1 < Hl)
        s += (1.0f - lx) * ly * vbase[(size_t)((y0 + 1) * Wl + x0) * DMODEL];
      if (x0 + 1 >= 0 && x0 + 1 < Wl && y0 + 1 >= 0 && y0 + 1 < Hl)
        s += lx * ly * vbase[(size_t)((y0 + 1) * Wl + x0 + 1) * DMODEL];
      a += (w[l * NPOINTS + p] * inv) * s;
    }
  }
  acc[(size_t)nq * DMODEL + h * DHEAD + lane] = f2bf(a);  // [n,q,h,dh] contiguous
}

extern "C" void kernel_launch(void* const* d_in, const int* in_sizes, int n_in,
                              void* d_out, int out_size, void* d_ws, size_t ws_size,
                              hipStream_t stream) {
  const float* query = (const float*)d_in[0];
  const float* refp  = (const float*)d_in[1];
  const float* xin   = (const float*)d_in[2];
  const int*   shapes= (const int*)d_in[3];
  const int*   lsi   = (const int*)d_in[4];
  const float* Wsamp = (const float*)d_in[5];
  const float* bsamp = (const float*)d_in[6];
  const float* Wattn = (const float*)d_in[7];
  const float* battn = (const float*)d_in[8];
  const float* Wval  = (const float*)d_in[9];
  const float* bval  = (const float*)d_in[10];
  const float* Wout  = (const float*)d_in[11];
  const float* bout  = (const float*)d_in[12];
  float* out = (float*)d_out;

  const int Lq = in_sizes[0] / (BATCH * DMODEL);   // 19947
  const int M  = BATCH * Lq;                       // 39894 GEMM rows

  // scratch layout: f32 intermediates first, then bf16 operands
  char* p = (char*)d_ws;
  float* value = (float*)p;  p += (size_t)M * 256 * sizeof(float);
  float* offr  = (float*)p;  p += (size_t)M * 256 * sizeof(float);
  float* attr  = (float*)p;  p += (size_t)M * 128 * sizeof(float);
  __bf16* Xbf  = (__bf16*)p; p += (size_t)M * 256 * sizeof(__bf16);
  __bf16* Qbf  = (__bf16*)p; p += (size_t)M * 256 * sizeof(__bf16);
  __bf16* Abf  = (__bf16*)p; p += (size_t)M * 256 * sizeof(__bf16);   // sampler acc
  __bf16* Wvbf = (__bf16*)p; p += (size_t)256 * 256 * sizeof(__bf16);
  __bf16* Wsbf = (__bf16*)p; p += (size_t)256 * 256 * sizeof(__bf16);
  __bf16* Wabf = (__bf16*)p; p += (size_t)128 * 256 * sizeof(__bf16);
  __bf16* Wobf = (__bf16*)p; p += (size_t)256 * 256 * sizeof(__bf16);

  auto cvt = [&](const float* s, __bf16* d, size_t n) {
    const int n8 = (int)(n / 8);                       // all sizes are multiples of 8
    cvt_f32_bf16<<<(n8 + 255) / 256, 256, 0, stream>>>(s, d, n8);
  };
  cvt(xin,   Xbf,  (size_t)M * 256);
  cvt(query, Qbf,  (size_t)M * 256);
  cvt(Wval,  Wvbf, 256 * 256);
  cvt(Wsamp, Wsbf, 256 * 256);
  cvt(Wattn, Wabf, 128 * 256);
  cvt(Wout,  Wobf, 256 * 256);

  const int WPB = 4;          // waves per block (blockDim = 128)
  auto gemm = [&](const __bf16* X, const __bf16* W, const float* b, float* Y, int Nout) {
    const int tiles = ((M + 15) / 16) * (Nout / 64);   // one wave per 16x64 strip
    const int blocks = (tiles + WPB - 1) / WPB;
    gemm256_bf16_wmma<<<blocks, WPB * 32, 0, stream>>>(X, W, b, Y, M, Nout);
  };

  gemm(Xbf, Wvbf, bval,  value, 256);      // value projection
  gemm(Qbf, Wsbf, bsamp, offr,  256);      // sampling offsets
  gemm(Qbf, Wabf, battn, attr,  128);      // attention logits

  const int totalWaves = M * NHEADS;       // one wave per (n,q,h)
  const int sblocks = (totalWaves + 7) / 8;
  msda_sample<<<sblocks, 256, 0, stream>>>(refp, shapes, lsi, value, offr, attr, Abf, Lq);

  gemm(Abf, Wobf, bout, out, 256);         // output projection -> d_out
}
